// FaceTokenVQ_4664334483523
// MI455X (gfx1250) — compile-verified, matrix operand order checked by reading, not run on hardware
//
#include <hip/hip_runtime.h>

// ---------------------------------------------------------------------------
// FaceTokenVQ on MI455X (gfx1250, wave32, WMMA)
//
// rows = B*H = 65536 (width D*K = 256), N = 4096 codes.
// Dominant cost: argmin_n ||x_r - cb_n||^2 -> X(65536x256) @ CB^T(256x4096),
// 137 GFLOP. Computed with f16 hi/lo split precision (hi*hi + hi*lo + lo*hi)
// on v_wmma_f32_16x16x32_f16 (~2^-22 rel err, argmin-safe), ||x||^2 dropped
// (row-constant), argmin fused into the WMMA epilogue.
//
// Round-2 change: the round-1 loop stalled on B loads (24 FLOP/byte of B =>
// ~100 TB/s L2 demand at WMMA peak). Now each wave caches 32 rows of A in
// VGPRs (2 row tiles, 256 VGPRs) and drives TWO independent accumulator
// chains per B chunk: 6 wmma per 64B-of-loads/lane, halving B-feed bandwidth
// and doubling WMMA ILP.
// ---------------------------------------------------------------------------

typedef _Float16 f16;
typedef __attribute__((ext_vector_type(8)))  _Float16 v8h;
typedef __attribute__((ext_vector_type(16))) _Float16 v16h;
typedef __attribute__((ext_vector_type(8)))  float    v8f;
typedef __attribute__((ext_vector_type(8)))  float    f32x8;

#define SHUF16(a,b) __builtin_shufflevector(a, b, 0,1,2,3,4,5,6,7,8,9,10,11,12,13,14,15)
#define WMMA_F16(A,B,C) __builtin_amdgcn_wmma_f32_16x16x32_f16(false, A, false, B, (short)0, C, false, false)

static __device__ __forceinline__ float gelu_tanh(float v) {
  // jax.nn.gelu default (approximate=True, tanh form)
  const float c = 0.7978845608028654f;   // sqrt(2/pi)
  return 0.5f * v * (1.0f + tanhf(c * (v + 0.044715f * v * v * v)));
}

static __device__ __forceinline__ float wred32(float v) {
  #pragma unroll
  for (int m = 16; m >= 1; m >>= 1) v += __shfl_xor(v, m, 32);
  return v;
}

// ---------------------------------------------------------------------------
// Kernel A: cb = W_o @ ( (W_i @ fc[n]) * gelu(scalar) ), per code n.
// One 256-thread block per code; emits cb fp32, f16 hi/lo split, and ||cb||^2.
// ---------------------------------------------------------------------------
__global__ void vq_build_codebook(const float* __restrict__ fc,
                                  const float* __restrict__ Wi,
                                  const float* __restrict__ Wo,
                                  float* __restrict__ cb,
                                  f16*   __restrict__ cb_hi,
                                  f16*   __restrict__ cb_lo,
                                  float* __restrict__ cbn2) {
  __shared__ float sFc[256], sWi[256], sWo[256], sH[256], sG[256], sRed[256];
  const int n = blockIdx.x;
  const int t = threadIdx.x;
  sWi[t] = Wi[t];
  sWo[t] = Wo[t];
  sFc[t] = fc[n * 256 + t];
  __syncthreads();

  const int o = t >> 4;      // output channel
  const int k = t & 15;      // blade component
  float h = 0.0f;
  #pragma unroll
  for (int i = 0; i < 16; ++i) h += sWi[o * 16 + i] * sFc[i * 16 + k];
  sH[t] = h;
  __syncthreads();

  const float g = gelu_tanh(sH[o * 16]);   // gate by scalar (k=0) component
  sG[t] = h * g;
  __syncthreads();

  float c = 0.0f;
  #pragma unroll
  for (int d = 0; d < 16; ++d) c += sWo[o * 16 + d] * sG[d * 16 + k];

  const int flat = n * 256 + t;
  cb[flat] = c;
  const f16 chi = (f16)c;
  cb_hi[flat] = chi;
  cb_lo[flat] = (f16)(c - (float)chi);

  sRed[t] = c * c;
  __syncthreads();
  #pragma unroll
  for (int s = 128; s > 0; s >>= 1) {
    if (t < s) sRed[t] += sRed[t + s];
    __syncthreads();
  }
  if (t == 0) cbn2[n] = sRed[0];
}

// ---------------------------------------------------------------------------
// Kernel B: fused WMMA distance + argmin, M=32 register blocking.
// One wave = two 16-row tiles of X vs all 4096 codes. A tiles live in VGPRs
// (2 x 8 k-chunks x {hi,lo} v16h = 256 VGPRs). Per code tile: 8 chunks x
// 6 wmma off one B-chunk load -> 48 v_wmma static, two independent C chains.
// 16-bit operand striping per ISA 7.12.2: lane L (g = L>>4) half-element j
// holds K = 32c + 16*(j>>3) + 8*g + (j&7); A lanes index M, B lanes index N.
// C/D: lane L reg r holds (M = r + 8*(L>>4), N = L&15).
// ---------------------------------------------------------------------------
__global__ __launch_bounds__(128)
void vq_wmma_argmin(const float* __restrict__ x,
                    const f16*   __restrict__ cb_hi,
                    const f16*   __restrict__ cb_lo,
                    const float* __restrict__ cbn2,
                    int*         __restrict__ idxOut) {
  const int wave = (blockIdx.x << 2) + (threadIdx.x >> 5);  // 0..2047
  const int lane = threadIdx.x & 31;
  const int ln   = lane & 15;   // M within tile for A-load; N for B/epilogue
  const int g    = lane >> 4;   // K-group select

  // ---- Load & split two 16x256 fp32 A tiles into f16 hi/lo operands ----
  const float* xrow0 = x + (size_t)(wave * 32 + ln) * 256;
  const float* xrow1 = xrow0 + 16 * 256;
  v16h a0h[8], a0l[8], a1h[8], a1l[8];
  #pragma unroll
  for (int c = 0; c < 8; ++c) {
    {
      const f32x8 s0 = *(const f32x8*)(xrow0 + 32 * c + 8 * g);
      const f32x8 s1 = *(const f32x8*)(xrow0 + 32 * c + 16 + 8 * g);
      const v8h  h0 = __builtin_convertvector(s0, v8h);
      const v8h  h1 = __builtin_convertvector(s1, v8h);
      const v8h  l0 = __builtin_convertvector(s0 - __builtin_convertvector(h0, f32x8), v8h);
      const v8h  l1 = __builtin_convertvector(s1 - __builtin_convertvector(h1, f32x8), v8h);
      a0h[c] = SHUF16(h0, h1);
      a0l[c] = SHUF16(l0, l1);
    }
    {
      const f32x8 s0 = *(const f32x8*)(xrow1 + 32 * c + 8 * g);
      const f32x8 s1 = *(const f32x8*)(xrow1 + 32 * c + 16 + 8 * g);
      const v8h  h0 = __builtin_convertvector(s0, v8h);
      const v8h  h1 = __builtin_convertvector(s1, v8h);
      const v8h  l0 = __builtin_convertvector(s0 - __builtin_convertvector(h0, f32x8), v8h);
      const v8h  l1 = __builtin_convertvector(s1 - __builtin_convertvector(h1, f32x8), v8h);
      a1h[c] = SHUF16(h0, h1);
      a1l[c] = SHUF16(l0, l1);
    }
  }

  float minv0[8], minv1[8];
  int   mini0[8], mini1[8];
  #pragma unroll
  for (int r = 0; r < 8; ++r) {
    minv0[r] = 3.4e38f; mini0[r] = 0;
    minv1[r] = 3.4e38f; mini1[r] = 0;
  }

  for (int nt = 0; nt < 256; ++nt) {
    const f16* crh = cb_hi + (size_t)(nt * 16 + ln) * 256;
    const f16* crl = cb_lo + (size_t)(nt * 16 + ln) * 256;
    __builtin_prefetch(crh + 16 * 256, 0, 1);   // next code tile
    __builtin_prefetch(crl + 16 * 256, 0, 1);

    v8f acc0 = {};
    v8f acc1 = {};
    #pragma unroll
    for (int c = 0; c < 8; ++c) {
      const v8h b0h = *(const v8h*)(crh + 32 * c + 8 * g);
      const v8h b1h = *(const v8h*)(crh + 32 * c + 16 + 8 * g);
      const v8h b0l = *(const v8h*)(crl + 32 * c + 8 * g);
      const v8h b1l = *(const v8h*)(crl + 32 * c + 16 + 8 * g);
      const v16h bh = SHUF16(b0h, b1h);
      const v16h bl = SHUF16(b0l, b1l);
      // Two independent accumulation chains share each loaded B chunk.
      acc0 = WMMA_F16(a0h[c], bh, acc0);
      acc1 = WMMA_F16(a1h[c], bh, acc1);
      acc0 = WMMA_F16(a0h[c], bl, acc0);
      acc1 = WMMA_F16(a1h[c], bl, acc1);
      acc0 = WMMA_F16(a0l[c], bh, acc0);
      acc1 = WMMA_F16(a1l[c], bh, acc1);
    }

    const int   code = nt * 16 + ln;               // this lane's N column
    const float cn2  = cbn2[code];
    #pragma unroll
    for (int r = 0; r < 8; ++r) {
      const float d0 = cn2 - 2.0f * acc0[r];       // ||x||^2 dropped (row-const)
      const float d1 = cn2 - 2.0f * acc1[r];
      if (d0 < minv0[r]) { minv0[r] = d0; mini0[r] = code; }
      if (d1 < minv1[r]) { minv1[r] = d1; mini1[r] = code; }
    }
  }

  // ---- reduce min over the N dimension: 16-lane groups (masks 8..1) ----
  #pragma unroll
  for (int r = 0; r < 8; ++r) {
    #pragma unroll
    for (int mk = 8; mk >= 1; mk >>= 1) {
      {
        const float ov = __shfl_xor(minv0[r], mk, 32);
        const int   oi = __shfl_xor(mini0[r], mk, 32);
        if (ov < minv0[r] || (ov == minv0[r] && oi < mini0[r])) {
          minv0[r] = ov; mini0[r] = oi;
        }
      }
      {
        const float ov = __shfl_xor(minv1[r], mk, 32);
        const int   oi = __shfl_xor(mini1[r], mk, 32);
        if (ov < minv1[r] || (ov == minv1[r] && oi < mini1[r])) {
          minv1[r] = ov; mini1[r] = oi;
        }
      }
    }
  }
  if (ln == 0) {                      // lane 0 -> M 0..7, lane 16 -> M 8..15
    #pragma unroll
    for (int r = 0; r < 8; ++r) {
      idxOut[wave * 32 + g * 8 + r]      = mini0[r];
      idxOut[wave * 32 + 16 + g * 8 + r] = mini1[r];
    }
  }
}

// ---------------------------------------------------------------------------
// Kernel C: gather e = cb[i], Householder-rotation STE, write (e, s).
// One wave per 256-wide row, lane owns 8 contiguous floats (b128 pairs).
// Memory-bound: ~200MB => ~9us at 23.3 TB/s.
// ---------------------------------------------------------------------------
__global__ void vq_householder(const float* __restrict__ x,
                               const float* __restrict__ cb,
                               const int*   __restrict__ idx,
                               float* __restrict__ outE,
                               float* __restrict__ outS) {
  const int wave = threadIdx.x >> 5;
  const int lane = threadIdx.x & 31;
  const size_t row = (size_t)blockIdx.x * 8 + wave;        // 0..65535
  const float* xr = x  + row * 256 + lane * 8;
  const float* er = cb + (size_t)idx[row] * 256 + lane * 8;

  const f32x8 xf = *(const f32x8*)xr;   // lane's 32B slice of the row
  const f32x8 e  = *(const f32x8*)er;

  float xn2 = 0.0f, en2 = 0.0f;
  #pragma unroll
  for (int j = 0; j < 8; ++j) { xn2 += xf[j] * xf[j]; en2 += e[j] * e[j]; }
  xn2 = wred32(xn2);
  en2 = wred32(en2);
  const float xn = sqrtf(xn2), en = sqrtf(en2);
  const float rx = 1.0f / fmaxf(xn, 1e-6f);
  const float re = 1.0f / fmaxf(en, 1e-6f);

  f32x8 sdr;
  float sd2 = 0.0f;
  #pragma unroll
  for (int j = 0; j < 8; ++j) {
    sdr[j] = xf[j] * rx + e[j] * re;               // xd + ed
    sd2 += sdr[j] * sdr[j];
  }
  sd2 = wred32(sd2);
  const float rs = 1.0f / fmaxf(sqrtf(sd2), 1e-6f);

  float sdx = 0.0f, xdx = 0.0f;
  #pragma unroll
  for (int j = 0; j < 8; ++j) {
    sdx += (sdr[j] * rs) * xf[j];                  // sd . x
    xdx += (xf[j] * rx) * xf[j];                   // xd . x
  }
  sdx = wred32(sdx);
  xdx = wred32(xdx);

  const float scale = en * rx;                     // en / max(xn, eps)
  f32x8 sv;
  #pragma unroll
  for (int j = 0; j < 8; ++j) {
    const float sdj = sdr[j] * rs;
    const float edj = e[j] * re;
    const float r   = xf[j] - 2.0f * sdj * sdx + 2.0f * edj * xdx;
    sv[j] = r * scale;
  }
  *(f32x8*)(outE + row * 256 + lane * 8) = e;
  *(f32x8*)(outS + row * 256 + lane * 8) = sv;
}

// ---------------------------------------------------------------------------
extern "C" void kernel_launch(void* const* d_in, const int* in_sizes, int n_in,
                              void* d_out, int out_size, void* d_ws, size_t ws_size,
                              hipStream_t stream) {
  const float* x  = (const float*)d_in[0];   // (8192, 128, 16)
  const float* fc = (const float*)d_in[1];   // (4096, 16, 16)
  const float* Wi = (const float*)d_in[2];   // (16, 16)
  const float* Wo = (const float*)d_in[3];   // (16, 16)

  char* ws = (char*)d_ws;                    // ~8.3 MiB used
  float* cb    = (float*)(ws);                                   // 4 MiB
  f16*   cb_hi = (f16*)  (ws + (4u << 20));                      // 2 MiB
  f16*   cb_lo = (f16*)  (ws + (6u << 20));                      // 2 MiB
  float* cbn2  = (float*)(ws + (8u << 20));                      // 16 KiB
  int*   idx   = (int*)  (ws + (8u << 20) + (64u << 10));        // 256 KiB

  float* outE = (float*)d_out;
  float* outS = outE + (size_t)8192 * 128 * 16;

  vq_build_codebook<<<4096, 256, 0, stream>>>(fc, Wi, Wo, cb, cb_hi, cb_lo, cbn2);
  vq_wmma_argmin   <<< 512, 128, 0, stream>>>(x, cb_hi, cb_lo, cbn2, idx);
  vq_householder   <<<8192, 256, 0, stream>>>(x, cb, idx, outE, outS);
}
